// WritheMessage_46411416601353
// MI455X (gfx1250) — matrix operand order repaired and checked by reading, not compiled, not run on hardware
//
#include <hip/hip_runtime.h>

#define BB 8          // batch
#define FF 64         // features
#define LEAKY 0.01f

typedef _Float16 v16h __attribute__((ext_vector_type(16)));
typedef _Float16 h8   __attribute__((ext_vector_type(8)));
typedef float    v8f  __attribute__((ext_vector_type(8)));

__device__ __forceinline__ float3 sub3(float3 a, float3 b){
    return make_float3(a.x-b.x, a.y-b.y, a.z-b.z);
}
__device__ __forceinline__ float dot3(float3 a, float3 b){
    return a.x*b.x + a.y*b.y + a.z*b.z;
}
__device__ __forceinline__ float3 cross3(float3 a, float3 b){
    return make_float3(a.y*b.z - a.z*b.y, a.z*b.x - a.x*b.z, a.x*b.y - a.y*b.x);
}
__device__ __forceinline__ float3 nrm3(float3 v){
    float r = rsqrtf(dot3(v, v));
    return make_float3(v.x*r, v.y*r, v.z*r);
}
__device__ __forceinline__ float3 ld3(const float* p){
    return make_float3(p[0], p[1], p[2]);
}
__device__ __forceinline__ float clamp1(float v){
    return fminf(1.0f, fmaxf(-1.0f, v));
}
// f16 p[0..7] and p[16..23] from LDS -> one WMMA fragment (two b128 loads)
__device__ __forceinline__ v16h ldsfrag(const _Float16* p){
    h8 lo = *(const h8*)(p);
    h8 hi = *(const h8*)(p + 16);
    return __builtin_shufflevector(lo, hi,
        0,1,2,3,4,5,6,7,8,9,10,11,12,13,14,15);
}

// ---------------- Stage 1: writhe scalar per (segment, batch) -------------
__global__ void writhe_kernel(const float* __restrict__ xyz,
                              const int*   __restrict__ seg,
                              float* __restrict__ wr, int P, int NA)
{
    int tid = blockIdx.x * blockDim.x + threadIdx.x;
    if (tid >= P * BB) return;
    int p = tid >> 3, b = tid & 7;

    const int* s = seg + 4 * p;
    const float* base = xyz + (size_t)b * NA * 3;
    float3 s0 = ld3(base + 3 * s[0]);
    float3 s1 = ld3(base + 3 * s[1]);
    float3 s2 = ld3(base + 3 * s[2]);
    float3 s3 = ld3(base + 3 * s[3]);

    float3 d0 = nrm3(sub3(s2, s0));
    float3 d1 = nrm3(sub3(s3, s0));
    float3 d2 = nrm3(sub3(s2, s1));
    float3 d3 = nrm3(sub3(s3, s1));

    float3 c0 = nrm3(cross3(d0, d1));
    float3 c1 = nrm3(cross3(d1, d3));
    float3 c2 = nrm3(cross3(d3, d2));
    float3 c3 = nrm3(cross3(d2, d0));

    float omega = asinf(clamp1(dot3(c0, c1)))
                + asinf(clamp1(dot3(c1, c3)))
                + asinf(clamp1(dot3(c2, c2)))
                + asinf(clamp1(dot3(c3, c0)));

    float sv = dot3(cross3(nrm3(sub3(s3, s2)), nrm3(sub3(s1, s0))), d0);
    float sgn = (sv > 0.0f ? 1.0f : 0.0f) - (sv < 0.0f ? 1.0f : 0.0f);

    wr[p * BB + b] = omega * sgn * 0.15915494309189535f; // 1/(2*pi)
}

// ------------- Stage 2: fused RBF -> W1 -> leaky -> W2 -> scatter ---------
// One wave handles 2 segments x 8 batches = 16 rows per iteration.
// Weights live in LDS (col-major f16); B-fragments reloaded per iteration
// to keep the wave under 256 VGPRs.
__global__ __launch_bounds__(256, 1)
void msg_kernel(const float* __restrict__ x,
                const int*   __restrict__ seg,
                const float* __restrict__ wr,
                const float* __restrict__ W1, const float* __restrict__ b1,
                const float* __restrict__ W2, const float* __restrict__ b2,
                float* __restrict__ out, int P, int NA)
{
    const int lane = threadIdx.x & 31;
    const int wid  = threadIdx.x >> 5;
    const int g    = lane >> 4;     // lane group
    const int n    = lane & 15;     // A row / B,C column within a 16-tile

    // ---- LDS: transposed weights (col-major) + per-wave h staging ----
    __shared__ _Float16 w1t[FF * FF];      // w1t[col*64 + k] = W1[k][col]
    __shared__ _Float16 w2t[FF * FF];
    __shared__ _Float16 hbuf[8][16 * 64];  // per-wave C->A relayout buffer

    for (int i = threadIdx.x; i < FF * FF; i += blockDim.x) {
        int k = i >> 6, col = i & 63;           // coalesced global read
        w1t[col * 64 + k] = (_Float16)W1[i];
        w2t[col * 64 + k] = (_Float16)W2[i];
    }
    __syncthreads();

    float bias1[4], bias2[4];
#pragma unroll
    for (int t = 0; t < 4; ++t) {
        bias1[t] = b1[16 * t + n];
        bias2[t] = b2[16 * t + n];
    }

    _Float16* hb = hbuf[wid];
    const int colbase = n * 64 + 8 * g;    // B-frag base: (16t+n)*64 handled per t

    const int nPairs     = (P + 1) >> 1;
    const int wavesTotal = gridDim.x * (blockDim.x >> 5);
    const int wgid       = blockIdx.x * (blockDim.x >> 5) + wid;

    for (int pair = wgid; pair < nPairs; pair += wavesTotal) {
        const int p0 = pair * 2;

        // this lane's A-row: row m = n  ->  segment p0+(n>>3), batch n&7
        const int pm = p0 + (n >> 3);
        const int bm = n & 7;
        const float w = (pm < P) ? wr[pm * BB + bm] : 0.0f;

        // ---- RBF features directly in WMMA A layout ----
        // halves e -> K = (e<8 ? e : e+8) + 8*g   (+32 for second k-step)
        v16h a0, a1;
#pragma unroll
        for (int e = 0; e < 16; ++e) {
            int k0 = (e < 8 ? e : e + 8) + 8 * g;
            float v0 = -1.0f + (2.0f / 63.0f) * (float)k0;
            float q0 = (w - v0) * 31.5f;
            a0[e] = (_Float16)(__expf(-q0 * q0) * (1.0f / 1.12f));
            float v1 = -1.0f + (2.0f / 63.0f) * (float)(k0 + 32);
            float q1 = (w - v1) * 31.5f;
            a1[e] = (_Float16)(__expf(-q1 * q1) * (1.0f / 1.12f));
        }

        // ---- GEMM 1: h = leaky_relu(rbf @ W1 + b1) ----
#pragma unroll
        for (int t = 0; t < 4; ++t) {
            const _Float16* wp = w1t + t * (16 * 64) + colbase;
            v16h bf0 = ldsfrag(wp);        // K = 8g..8g+7, 16+8g..16+8g+7
            v16h bf1 = ldsfrag(wp + 32);   // same, +32
            v8f c;
#pragma unroll
            for (int r = 0; r < 8; ++r) c[r] = bias1[t];
            c = __builtin_amdgcn_wmma_f32_16x16x32_f16(false, a0, false, bf0,
                                                       (short)0, c, false, false);
            c = __builtin_amdgcn_wmma_f32_16x16x32_f16(false, a1, false, bf1,
                                                       (short)0, c, false, false);
            // leaky relu + stage to LDS (C layout: col = 16t+n, row = r+8g)
#pragma unroll
            for (int r = 0; r < 8; ++r) {
                float hv = c[r] > 0.0f ? c[r] : LEAKY * c[r];
                hb[(r + 8 * g) * 64 + 16 * t + n] = (_Float16)hv;
            }
        }

        asm volatile("s_wait_dscnt 0" ::: "memory");

        // ---- read h back as A fragments (row m=n) ----
        h8 lo0 = *(const h8*)(hb + n * 64 + 8 * g);
        h8 hi0 = *(const h8*)(hb + n * 64 + 8 * g + 16);
        h8 lo1 = *(const h8*)(hb + n * 64 + 8 * g + 32);
        h8 hi1 = *(const h8*)(hb + n * 64 + 8 * g + 48);
        v16h ha0 = __builtin_shufflevector(lo0, hi0, 0,1,2,3,4,5,6,7,8,9,10,11,12,13,14,15);
        v16h ha1 = __builtin_shufflevector(lo1, hi1, 0,1,2,3,4,5,6,7,8,9,10,11,12,13,14,15);

        // ---- GEMM 2: gmsg = h @ W2 + b2 ----
        v8f d[4];
#pragma unroll
        for (int t = 0; t < 4; ++t) {
            const _Float16* wp = w2t + t * (16 * 64) + colbase;
            v16h bf0 = ldsfrag(wp);
            v16h bf1 = ldsfrag(wp + 32);
            v8f c;
#pragma unroll
            for (int r = 0; r < 8; ++r) c[r] = bias2[t];
            c = __builtin_amdgcn_wmma_f32_16x16x32_f16(false, ha0, false, bf0,
                                                       (short)0, c, false, false);
            c = __builtin_amdgcn_wmma_f32_16x16x32_f16(false, ha1, false, bf1,
                                                       (short)0, c, false, false);
            d[t] = c;
        }

        // ---- gather x, scatter both edge directions via f32 atomics ----
#pragma unroll
        for (int r = 0; r < 8; ++r) {
            const int row = r + 8 * g;
            const int p   = p0 + (row >> 3);
            if (p < P) {
                const int bb = row & 7;
                const int ai = seg[4 * p + 0];
                const int aj = seg[4 * p + 2];
                const float* xb = x   + (size_t)bb * NA * FF;
                float*       ob = out + (size_t)bb * NA * FF;
#pragma unroll
                for (int t = 0; t < 4; ++t) {
                    const int col = 16 * t + n;
                    const float gv = d[t][r];
                    atomicAdd(ob + (size_t)aj * FF + col, gv * xb[(size_t)ai * FF + col]);
                    atomicAdd(ob + (size_t)ai * FF + col, gv * xb[(size_t)aj * FF + col]);
                }
            }
        }
    }
}

extern "C" void kernel_launch(void* const* d_in, const int* in_sizes, int n_in,
                              void* d_out, int out_size, void* d_ws, size_t ws_size,
                              hipStream_t stream)
{
    const float* x   = (const float*)d_in[0];
    const float* xyz = (const float*)d_in[1];
    const int*   seg = (const int*)d_in[2];
    const float* W1  = (const float*)d_in[3];
    const float* b1  = (const float*)d_in[4];
    const float* W2  = (const float*)d_in[5];
    const float* b2  = (const float*)d_in[6];
    float* out = (float*)d_out;

    const int P  = in_sizes[2] / 4;
    const int NA = in_sizes[1] / (BB * 3);
    float* wr = (float*)d_ws;   // P*8 floats of scratch

    // out = x (atomics accumulate the aggregation on top)
    hipMemcpyAsync(out, x, (size_t)out_size * sizeof(float),
                   hipMemcpyDeviceToDevice, stream);

    const int totalW = P * BB;
    writhe_kernel<<<(totalW + 255) / 256, 256, 0, stream>>>(xyz, seg, wr, P, NA);

    // 1024 blocks x 8 waves = 8192 waves grid-striding over segment pairs
    msg_kernel<<<1024, 256, 0, stream>>>(x, seg, wr, W1, b1, W2, b2, out, P, NA);
}
